// MoETransformersBlock_22574348108132
// MI455X (gfx1250) — matrix operand order repaired and checked by reading, not compiled, not run on hardware
//
#include <hip/hip_runtime.h>
#include <hip/hip_bf16.h>

// ---------------------------------------------------------------------------
// MoE transformer block for gfx1250 (MI455X). wave32, WMMA bf16 everywhere.
// ---------------------------------------------------------------------------

typedef __attribute__((ext_vector_type(16))) __bf16 bf16x16;
typedef __attribute__((ext_vector_type(8)))  float  f32x8;

typedef unsigned short u16;
typedef unsigned int   u32;

// Problem constants (match reference)
#define BB   2
#define SS   2048
#define DD   1024
#define HH   16
#define GG   4
#define HDD  64
#define EE   8
#define HID  1024
#define TT   (BB*SS)          // 4096 tokens
#define TOPK 2

// GEMM tiling
#define BM 128
#define BN 64
#define BK 32
#define SA 40                  // LDS stride (bf16 elems) for A tiles (80B = 5*16B aligned)
#define SB 40                  // LDS stride for B tiles (stored transposed, [n][k])

// ---------------------------------------------------------------------------
// Helpers
// ---------------------------------------------------------------------------
__device__ __forceinline__ u16 f2bf(float f) {
  u32 u = __builtin_bit_cast(u32, f);
  u32 r = u + 0x7fffu + ((u >> 16) & 1u);   // round-to-nearest-even
  return (u16)(r >> 16);
}
__device__ __forceinline__ float bf2f(u16 h) {
  return __builtin_bit_cast(float, (u32)h << 16);
}

struct FragU8 { u32 v[8]; };

__device__ __forceinline__ f32x8 wmma_bf16(const u32* a, const u32* b, f32x8 c) {
  FragU8 ua, ub;
#pragma unroll
  for (int i = 0; i < 8; ++i) { ua.v[i] = a[i]; ub.v[i] = b[i]; }
  bf16x16 av = __builtin_bit_cast(bf16x16, ua);
  bf16x16 bv = __builtin_bit_cast(bf16x16, ub);
  return __builtin_amdgcn_wmma_f32_16x16x32_bf16(false, av, false, bv,
                                                 (short)0, c, false, false);
}

// A fragment (16x32 bf16, M x K): lane 0-15 -> M=lane, K 0-15; lane 16-31 -> K 16-31
__device__ __forceinline__ void load_a_frag(const u16* p, int stride, int lane, u32* u) {
  const int half = lane >> 4;
  const u16* row = p + (lane & 15) * stride;
#pragma unroll
  for (int v = 0; v < 8; ++v) {
    int k0 = (v << 1) + (half << 4);
    u[v] = *(const u32*)(row + k0);
  }
}
// B fragment (32x16 bf16, K x N) from LDS stored as [n][k]: lane 0-15 -> K 0-15, lane 16-31 -> K 16-31
__device__ __forceinline__ void load_b_frag(const u16* p, int stride, int lane, u32* u) {
  const int half = lane >> 4;
  const u16* row = p + (lane & 15) * stride;
#pragma unroll
  for (int v = 0; v < 8; ++v) {
    int k0 = (v << 1) + (half << 4);
    u[v] = *(const u32*)(row + k0);
  }
}

// ---------------------------------------------------------------------------
// K1/K6: rmsnorm (fp32 in -> bf16 out), one block per token row of 1024
// ---------------------------------------------------------------------------
__global__ void rmsnorm_kernel(const float* __restrict__ X,
                               const float* __restrict__ w,
                               u16* __restrict__ out) {
  __shared__ float red[256];
  const int t = blockIdx.x, tid = threadIdx.x;
  float4 v = ((const float4*)(X + (size_t)t * DD))[tid];
  float ss = v.x * v.x + v.y * v.y + v.z * v.z + v.w * v.w;
  red[tid] = ss; __syncthreads();
  for (int o = 128; o > 0; o >>= 1) {
    if (tid < o) red[tid] += red[tid + o];
    __syncthreads();
  }
  float scale = 1.0f / (sqrtf(red[0] / (float)DD) + 1e-6f);
  __syncthreads();
  float4 wv = ((const float4*)w)[tid];
  u16* o4 = out + (size_t)t * DD + tid * 4;
  o4[0] = f2bf(v.x * scale * wv.x);
  o4[1] = f2bf(v.y * scale * wv.y);
  o4[2] = f2bf(v.z * scale * wv.z);
  o4[3] = f2bf(v.w * scale * wv.w);
}

// ---------------------------------------------------------------------------
// Generic WMMA GEMM: Out[M,N] = A_bf16[M,K] * W_f32[K,N] (+ Res), fp32 out
// ---------------------------------------------------------------------------
__global__ void gemm_aw(const u16* __restrict__ A, const float* __restrict__ W,
                        float* __restrict__ Out, int M, int N, int K,
                        const float* __restrict__ Res) {
  __shared__ u16 sA[BM * SA];
  __shared__ u16 sB[BN * SB];
  const int tid = threadIdx.x, lane = tid & 31, wid = tid >> 5;
  const int half = lane >> 4, mm = lane & 15;
  const int waveM = wid >> 1, waveN = wid & 1;
  const int m0 = blockIdx.y * BM, n0 = blockIdx.x * BN;

  f32x8 z = {0.f,0.f,0.f,0.f,0.f,0.f,0.f,0.f};
  f32x8 acc[2][2];
#pragma unroll
  for (int i = 0; i < 2; ++i)
#pragma unroll
    for (int j = 0; j < 2; ++j) acc[i][j] = z;

  for (int kt = 0; kt < K; kt += BK) {
#pragma unroll
    for (int i = 0; i < 2; ++i) {               // A: 128x32 bf16
      int c = tid * 2 + i, row = c >> 2, col8 = (c & 3) << 3;
      uint4 val = make_uint4(0u, 0u, 0u, 0u);
      if (m0 + row < M)
        val = *(const uint4*)(A + (size_t)(m0 + row) * K + kt + col8);
      *(uint4*)&sA[row * SA + col8] = val;
    }
#pragma unroll
    for (int i = 0; i < 2; ++i) {               // B: 32x64 fp32 -> bf16, transposed
      int c = tid * 2 + i, k = c >> 4, n4 = (c & 15) << 2;
      float4 wv = *(const float4*)(W + (size_t)(kt + k) * N + n0 + n4);
      sB[(n4 + 0) * SB + k] = f2bf(wv.x);
      sB[(n4 + 1) * SB + k] = f2bf(wv.y);
      sB[(n4 + 2) * SB + k] = f2bf(wv.z);
      sB[(n4 + 3) * SB + k] = f2bf(wv.w);
    }
    __syncthreads();
    u32 af[2][8], bfr[2][8];
#pragma unroll
    for (int tm = 0; tm < 2; ++tm)
      load_a_frag(&sA[(waveM * 32 + tm * 16) * SA], SA, lane, af[tm]);
#pragma unroll
    for (int tn = 0; tn < 2; ++tn)
      load_b_frag(&sB[(waveN * 32 + tn * 16) * SB], SB, lane, bfr[tn]);
#pragma unroll
    for (int tm = 0; tm < 2; ++tm)
#pragma unroll
      for (int tn = 0; tn < 2; ++tn)
        acc[tm][tn] = wmma_bf16(af[tm], bfr[tn], acc[tm][tn]);
    __syncthreads();
  }
#pragma unroll
  for (int tm = 0; tm < 2; ++tm)
#pragma unroll
    for (int tn = 0; tn < 2; ++tn)
#pragma unroll
      for (int r = 0; r < 8; ++r) {
        int m = m0 + waveM * 32 + tm * 16 + r + half * 8;
        int n = n0 + waveN * 32 + tn * 16 + mm;
        if (m < M) {
          float v = acc[tm][tn][r];
          if (Res) v += Res[(size_t)m * N + n];
          Out[(size_t)m * N + n] = v;
        }
      }
}

// ---------------------------------------------------------------------------
// K3: per-head q/k rmsnorm + RoPE; pack Q,K bf16; V transposed bf16
// grid: (24 heads, T tokens), block: 32 (one wave)
// ---------------------------------------------------------------------------
__global__ void rope_pack_kernel(const float* __restrict__ Qf,
                                 const float* __restrict__ Kf,
                                 const float* __restrict__ Vf,
                                 const float* __restrict__ qw,
                                 const float* __restrict__ kw,
                                 const int* __restrict__ pos_ids,
                                 u16* __restrict__ Qbf, u16* __restrict__ Kbf,
                                 u16* __restrict__ Vt) {
  const int hh = blockIdx.x, t = blockIdx.y, lane = threadIdx.x;
  const int b = t / SS, s = t % SS;
  if (hh < HH + GG) {                          // Q or K: rmsnorm + rope
    const float* src; const float* w;
    if (hh < HH) { src = Qf + (size_t)t * (HH * HDD) + hh * HDD; w = qw; }
    else         { src = Kf + (size_t)t * (GG * HDD) + (hh - HH) * HDD; w = kw; }
    float x1 = src[lane], x2 = src[lane + 32];
    float ss2 = x1 * x1 + x2 * x2;
#pragma unroll
    for (int o = 16; o > 0; o >>= 1) ss2 += __shfl_xor(ss2, o, 32);
    float scale = 1.0f / (sqrtf(ss2 / (float)HDD) + 1e-6f);
    x1 = x1 * scale * w[lane];
    x2 = x2 * scale * w[lane + 32];
    float pos  = (float)pos_ids[t];
    float invf = __powf(10000.f, -(float)lane / 32.f);
    float ang = pos * invf, c = __cosf(ang), sn = __sinf(ang);
    float o1 = x1 * c - x2 * sn, o2 = x2 * c + x1 * sn;
    u16* dst;
    if (hh < HH) dst = Qbf + (((size_t)(b * HH + hh)) * SS + s) * HDD;
    else         dst = Kbf + (((size_t)(b * GG + (hh - HH))) * SS + s) * HDD;
    dst[lane] = f2bf(o1); dst[lane + 32] = f2bf(o2);
  } else {                                     // V: just convert + transpose
    const int g = hh - HH - GG;
    const float* src = Vf + (size_t)t * (GG * HDD) + g * HDD;
    size_t base = ((size_t)(b * GG + g)) * HDD * SS;
    Vt[base + (size_t)lane * SS + s]        = f2bf(src[lane]);
    Vt[base + (size_t)(lane + 32) * SS + s] = f2bf(src[lane + 32]);
  }
}

// ---------------------------------------------------------------------------
// K4: causal flash attention (GQA). grid (S/128, B*H), block 256 (8 waves).
// Each wave: 16 query rows; WMMA for QK^T (bf16) and PV (bf16 via LDS).
// ---------------------------------------------------------------------------
__global__ void attn_kernel(const u16* __restrict__ Qbf, const u16* __restrict__ Kbf,
                            const u16* __restrict__ Vt, u16* __restrict__ ctx) {
  __shared__ u16 ldsP[8][16 * SA];
  const int tid = threadIdx.x, lane = tid & 31, wid = tid >> 5;
  const int half = lane >> 4, mm = lane & 15;
  const int qbase = blockIdx.x * 128 + wid * 16;
  const int b = blockIdx.y / HH, h = blockIdx.y % HH, g = h >> 2;

  const u16* Qrow = Qbf + (((size_t)(b * HH + h)) * SS + qbase) * HDD + mm * HDD;
  const u16* Kb   = Kbf + ((size_t)(b * GG + g)) * SS * HDD;
  const u16* Vb   = Vt  + ((size_t)(b * GG + g)) * HDD * SS;

  u32 qf[2][8];
#pragma unroll
  for (int ks = 0; ks < 2; ++ks)
#pragma unroll
    for (int v = 0; v < 8; ++v) {
      int k0 = (v << 1) + (half << 4) + ks * 32;
      qf[ks][v] = *(const u32*)(Qrow + k0);
    }

  f32x8 z = {0.f,0.f,0.f,0.f,0.f,0.f,0.f,0.f};
  f32x8 o[4]; 
#pragma unroll
  for (int i = 0; i < 4; ++i) o[i] = z;
  float mrow[8], lrow[8];
#pragma unroll
  for (int r = 0; r < 8; ++r) { mrow[r] = -1e30f; lrow[r] = 0.f; }

  const int ntiles = (qbase + 15) / 32 + 1;
  for (int j = 0; j < ntiles; ++j) {
    const int kb0 = j * 32;
    f32x8 sfr[2];
#pragma unroll
    for (int tn = 0; tn < 2; ++tn) {
      f32x8 s = z;
#pragma unroll
      for (int ks = 0; ks < 2; ++ks) {
        u32 kfr[8];
        const u16* krow = Kb + (size_t)(kb0 + tn * 16 + mm) * HDD + ks * 32;
#pragma unroll
        for (int v = 0; v < 8; ++v)
          kfr[v] = *(const u32*)(krow + (v << 1) + (half << 4));
        s = wmma_bf16(qf[ks], kfr, s);
      }
      sfr[tn] = s;
    }
    // scale + causal mask (C layout: row = r + 8*half, col = tn*16 + mm)
#pragma unroll
    for (int tn = 0; tn < 2; ++tn)
#pragma unroll
      for (int r = 0; r < 8; ++r) {
        int q = qbase + r + half * 8, key = kb0 + tn * 16 + mm;
        float val = sfr[tn][r] * 0.125f;
        if (key > q) val = -1e30f;
        sfr[tn][r] = val;
      }
    // online softmax stats (reduce across 16-lane half groups)
    float scl[8];
#pragma unroll
    for (int r = 0; r < 8; ++r) {
      float mx = fmaxf(sfr[0][r], sfr[1][r]);
#pragma unroll
      for (int off = 8; off > 0; off >>= 1) mx = fmaxf(mx, __shfl_xor(mx, off, 32));
      float mn = fmaxf(mrow[r], mx);
      scl[r] = __expf(mrow[r] - mn);
      mrow[r] = mn;
      float p0 = __expf(sfr[0][r] - mn), p1 = __expf(sfr[1][r] - mn);
      sfr[0][r] = p0; sfr[1][r] = p1;
      float rs = p0 + p1;
#pragma unroll
      for (int off = 8; off > 0; off >>= 1) rs += __shfl_xor(rs, off, 32);
      lrow[r] = lrow[r] * scl[r] + rs;
    }
#pragma unroll
    for (int nt = 0; nt < 4; ++nt)
#pragma unroll
      for (int r = 0; r < 8; ++r) o[nt][r] *= scl[r];
    // P -> LDS (wave-private) as bf16, reload as A fragment
    u16* P = &ldsP[wid][0];
#pragma unroll
    for (int tn = 0; tn < 2; ++tn)
#pragma unroll
      for (int r = 0; r < 8; ++r)
        P[(r + half * 8) * SA + tn * 16 + mm] = f2bf(sfr[tn][r]);
    u32 pf[8];
    load_a_frag(P, SA, lane, pf);
    // PV: B fragment direct from V^T (keys contiguous)
#pragma unroll
    for (int nt = 0; nt < 4; ++nt) {
      u32 vb[8];
      const u16* vrow = Vb + (size_t)(nt * 16 + mm) * SS + kb0;
#pragma unroll
      for (int v = 0; v < 8; ++v)
        vb[v] = *(const u32*)(vrow + (v << 1) + (half << 4));
      o[nt] = wmma_bf16(pf, vb, o[nt]);
    }
  }
  // epilogue: normalize by row sum and store ctx[T, H*HD] bf16
  float inv[8];
#pragma unroll
  for (int r = 0; r < 8; ++r) inv[r] = 1.0f / lrow[r];
#pragma unroll
  for (int nt = 0; nt < 4; ++nt)
#pragma unroll
    for (int r = 0; r < 8; ++r) {
      int qm = qbase + r + half * 8;
      ctx[((size_t)(b * SS + qm)) * (HH * HDD) + h * HDD + nt * 16 + mm] =
          f2bf(o[nt][r] * inv[r]);
    }
}

// ---------------------------------------------------------------------------
// K7: router — logits, top-2, softmax weights, per-expert counts
// ---------------------------------------------------------------------------
__global__ void router_kernel(const u16* __restrict__ h2, const float* __restrict__ gate_w,
                              int* __restrict__ tok_idx, float* __restrict__ tok_w,
                              int* __restrict__ counts) {
  const int lane = threadIdx.x & 31;
  const int t = blockIdx.x * 8 + (threadIdx.x >> 5);
  float acc[EE];
#pragma unroll
  for (int e = 0; e < EE; ++e) acc[e] = 0.f;
  for (int d = lane; d < DD; d += 32) {
    float xv = bf2f(h2[(size_t)t * DD + d]);
    const float* gw = gate_w + (size_t)d * EE;
#pragma unroll
    for (int e = 0; e < EE; ++e) acc[e] += xv * gw[e];
  }
#pragma unroll
  for (int e = 0; e < EE; ++e)
#pragma unroll
    for (int off = 16; off > 0; off >>= 1) acc[e] += __shfl_xor(acc[e], off, 32);
  if (lane == 0) {
    int i0 = 0; float v0 = acc[0];
#pragma unroll
    for (int e = 1; e < EE; ++e) if (acc[e] > v0) { v0 = acc[e]; i0 = e; }
    int i1 = -1; float v1 = -1e30f;
#pragma unroll
    for (int e = 0; e < EE; ++e) if (e != i0 && acc[e] > v1) { v1 = acc[e]; i1 = e; }
    float e1 = __expf(v1 - v0), ssum = 1.0f + e1;
    tok_idx[2 * t] = i0; tok_idx[2 * t + 1] = i1;
    tok_w[2 * t] = 1.0f / ssum; tok_w[2 * t + 1] = e1 / ssum;
    atomicAdd(&counts[i0], 1); atomicAdd(&counts[i1], 1);
  }
}

__global__ void zero_counts_kernel(int* counts) {
  if (threadIdx.x < EE) counts[threadIdx.x] = 0;
}
__global__ void bases_kernel(const int* counts, int* bases) {
  if (threadIdx.x == 0) {
    int s = 0;
#pragma unroll
    for (int e = 0; e < EE; ++e) { bases[e] = s; s += counts[e]; }
  }
}

// K9: deterministic per-expert token lists via block scan (grid = E)
__global__ void build_lists_kernel(const int* __restrict__ tok_idx,
                                   const float* __restrict__ tok_w,
                                   const int* __restrict__ bases,
                                   int* __restrict__ entry_token,
                                   float* __restrict__ entry_w,
                                   int* __restrict__ entry_of) {
  __shared__ int scan[256];
  __shared__ int running;
  const int e = blockIdx.x, tid = threadIdx.x;
  if (tid == 0) running = 0;
  __syncthreads();
  for (int c = 0; c < TT; c += 256) {
    int t = c + tid, slot = -1;
    if (tok_idx[2 * t] == e) slot = 0;
    else if (tok_idx[2 * t + 1] == e) slot = 1;
    int flag = (slot >= 0) ? 1 : 0;
    scan[tid] = flag; __syncthreads();
    for (int off = 1; off < 256; off <<= 1) {
      int v = (tid >= off) ? scan[tid - off] : 0;
      __syncthreads();
      scan[tid] += v;
      __syncthreads();
    }
    if (flag) {
      int g = bases[e] + running + scan[tid] - 1;
      entry_token[g] = t;
      entry_w[g] = tok_w[2 * t + slot];
      entry_of[2 * t + slot] = g;
    }
    __syncthreads();
    if (tid == 0) running += scan[255];
    __syncthreads();
  }
}

// ---------------------------------------------------------------------------
// K10: gathered gate/up GEMM + SiLU*up, bf16 out. grid (HID/BN, T*2/BM? ->32, E)
// ---------------------------------------------------------------------------
__global__ void gemm_gateup(const u16* __restrict__ A, const float* __restrict__ Wg,
                            const float* __restrict__ Wu, u16* __restrict__ He,
                            const int* __restrict__ entry_token,
                            const int* __restrict__ bases, const int* __restrict__ counts) {
  const int e = blockIdx.z;
  const int cnt = counts[e];
  const int m0 = blockIdx.y * BM;
  if (m0 >= cnt) return;
  const int base = bases[e];
  const float* WgE = Wg + (size_t)e * DD * HID;
  const float* WuE = Wu + (size_t)e * DD * HID;

  __shared__ u16 sA[BM * SA];
  __shared__ u16 sBg[BN * SB];
  __shared__ u16 sBu[BN * SB];
  const int tid = threadIdx.x, lane = tid & 31, wid = tid >> 5;
  const int half = lane >> 4, mm = lane & 15;
  const int waveM = wid >> 1, waveN = wid & 1;
  const int n0 = blockIdx.x * BN;

  f32x8 z = {0.f,0.f,0.f,0.f,0.f,0.f,0.f,0.f};
  f32x8 accg[2][2], accu[2][2];
#pragma unroll
  for (int i = 0; i < 2; ++i)
#pragma unroll
    for (int j = 0; j < 2; ++j) { accg[i][j] = z; accu[i][j] = z; }

  for (int kt = 0; kt < DD; kt += BK) {
#pragma unroll
    for (int i = 0; i < 2; ++i) {
      int c = tid * 2 + i, row = c >> 2, col8 = (c & 3) << 3;
      uint4 val = make_uint4(0u, 0u, 0u, 0u);
      if (m0 + row < cnt) {
        int tok = entry_token[base + m0 + row];
        val = *(const uint4*)(A + (size_t)tok * DD + kt + col8);
      }
      *(uint4*)&sA[row * SA + col8] = val;
    }
#pragma unroll
    for (int i = 0; i < 2; ++i) {
      int c = tid * 2 + i, k = c >> 4, n4 = (c & 15) << 2;
      float4 wg = *(const float4*)(WgE + (size_t)(kt + k) * HID + n0 + n4);
      float4 wu = *(const float4*)(WuE + (size_t)(kt + k) * HID + n0 + n4);
      sBg[(n4 + 0) * SB + k] = f2bf(wg.x); sBg[(n4 + 1) * SB + k] = f2bf(wg.y);
      sBg[(n4 + 2) * SB + k] = f2bf(wg.z); sBg[(n4 + 3) * SB + k] = f2bf(wg.w);
      sBu[(n4 + 0) * SB + k] = f2bf(wu.x); sBu[(n4 + 1) * SB + k] = f2bf(wu.y);
      sBu[(n4 + 2) * SB + k] = f2bf(wu.z); sBu[(n4 + 3) * SB + k] = f2bf(wu.w);
    }
    __syncthreads();
    u32 af[2][8], bg[2][8], bu[2][8];
#pragma unroll
    for (int tm = 0; tm < 2; ++tm)
      load_a_frag(&sA[(waveM * 32 + tm * 16) * SA], SA, lane, af[tm]);
#pragma unroll
    for (int tn = 0; tn < 2; ++tn) {
      load_b_frag(&sBg[(waveN * 32 + tn * 16) * SB], SB, lane, bg[tn]);
      load_b_frag(&sBu[(waveN * 32 + tn * 16) * SB], SB, lane, bu[tn]);
    }
#pragma unroll
    for (int tm = 0; tm < 2; ++tm)
#pragma unroll
      for (int tn = 0; tn < 2; ++tn) {
        accg[tm][tn] = wmma_bf16(af[tm], bg[tn], accg[tm][tn]);
        accu[tm][tn] = wmma_bf16(af[tm], bu[tn], accu[tm][tn]);
      }
    __syncthreads();
  }
#pragma unroll
  for (int tm = 0; tm < 2; ++tm)
#pragma unroll
    for (int tn = 0; tn < 2; ++tn)
#pragma unroll
      for (int r = 0; r < 8; ++r) {
        int m = m0 + waveM * 32 + tm * 16 + r + half * 8;
        int n = n0 + waveN * 32 + tn * 16 + mm;
        if (m < cnt) {
          float gv = accg[tm][tn][r], uv = accu[tm][tn][r];
          float hv = gv / (1.0f + __expf(-gv)) * uv;   // silu(g)*u
          He[(size_t)(base + m) * HID + n] = f2bf(hv);
        }
      }
}

// ---------------------------------------------------------------------------
// K11: per-expert down GEMM, scaled by routing weight, fp32 out
// ---------------------------------------------------------------------------
__global__ void gemm_down(const u16* __restrict__ He, const float* __restrict__ Wd,
                          float* __restrict__ DOut, const float* __restrict__ entry_w,
                          const int* __restrict__ bases, const int* __restrict__ counts) {
  const int e = blockIdx.z;
  const int cnt = counts[e];
  const int m0 = blockIdx.y * BM;
  if (m0 >= cnt) return;
  const int base = bases[e];
  const u16* A = He + (size_t)base * HID;
  const float* W = Wd + (size_t)e * HID * DD;

  __shared__ u16 sA[BM * SA];
  __shared__ u16 sB[BN * SB];
  const int tid = threadIdx.x, lane = tid & 31, wid = tid >> 5;
  const int half = lane >> 4, mm = lane & 15;
  const int waveM = wid >> 1, waveN = wid & 1;
  const int n0 = blockIdx.x * BN;

  f32x8 z = {0.f,0.f,0.f,0.f,0.f,0.f,0.f,0.f};
  f32x8 acc[2][2];
#pragma unroll
  for (int i = 0; i < 2; ++i)
#pragma unroll
    for (int j = 0; j < 2; ++j) acc[i][j] = z;

  for (int kt = 0; kt < HID; kt += BK) {
#pragma unroll
    for (int i = 0; i < 2; ++i) {
      int c = tid * 2 + i, row = c >> 2, col8 = (c & 3) << 3;
      uint4 val = make_uint4(0u, 0u, 0u, 0u);
      if (m0 + row < cnt)
        val = *(const uint4*)(A + (size_t)(m0 + row) * HID + kt + col8);
      *(uint4*)&sA[row * SA + col8] = val;
    }
#pragma unroll
    for (int i = 0; i < 2; ++i) {
      int c = tid * 2 + i, k = c >> 4, n4 = (c & 15) << 2;
      float4 wv = *(const float4*)(W + (size_t)(kt + k) * DD + n0 + n4);
      sB[(n4 + 0) * SB + k] = f2bf(wv.x); sB[(n4 + 1) * SB + k] = f2bf(wv.y);
      sB[(n4 + 2) * SB + k] = f2bf(wv.z); sB[(n4 + 3) * SB + k] = f2bf(wv.w);
    }
    __syncthreads();
    u32 af[2][8], bfr[2][8];
#pragma unroll
    for (int tm = 0; tm < 2; ++tm)
      load_a_frag(&sA[(waveM * 32 + tm * 16) * SA], SA, lane, af[tm]);
#pragma unroll
    for (int tn = 0; tn < 2; ++tn)
      load_b_frag(&sB[(waveN * 32 + tn * 16) * SB], SB, lane, bfr[tn]);
#pragma unroll
    for (int tm = 0; tm < 2; ++tm)
#pragma unroll
      for (int tn = 0; tn < 2; ++tn)
        acc[tm][tn] = wmma_bf16(af[tm], bfr[tn], acc[tm][tn]);
    __syncthreads();
  }
#pragma unroll
  for (int tm = 0; tm < 2; ++tm)
#pragma unroll
    for (int tn = 0; tn < 2; ++tn)
#pragma unroll
      for (int r = 0; r < 8; ++r) {
        int m = m0 + waveM * 32 + tm * 16 + r + half * 8;
        int n = n0 + waveN * 32 + tn * 16 + mm;
        if (m < cnt) {
          int g = base + m;
          DOut[(size_t)g * DD + n] = acc[tm][tn][r] * entry_w[g];
        }
      }
}

// K12: out = x2 + down(expert0) + down(expert1)
__global__ void combine_kernel(const float* __restrict__ x2, const float* __restrict__ DOut,
                               const int* __restrict__ entry_of, float* __restrict__ out) {
  const int t = blockIdx.x, tid = threadIdx.x;
  const int g0 = entry_of[2 * t], g1 = entry_of[2 * t + 1];
  float4 a = ((const float4*)(x2 + (size_t)t * DD))[tid];
  float4 d0 = ((const float4*)(DOut + (size_t)g0 * DD))[tid];
  float4 d1 = ((const float4*)(DOut + (size_t)g1 * DD))[tid];
  float4 r = make_float4(a.x + d0.x + d1.x, a.y + d0.y + d1.y,
                         a.z + d0.z + d1.z, a.w + d0.w + d1.w);
  ((float4*)(out + (size_t)t * DD))[tid] = r;
}

// ---------------------------------------------------------------------------
// Launch
// ---------------------------------------------------------------------------
extern "C" void kernel_launch(void* const* d_in, const int* in_sizes, int n_in,
                              void* d_out, int out_size, void* d_ws, size_t ws_size,
                              hipStream_t stream) {
  (void)in_sizes; (void)n_in; (void)out_size; (void)ws_size;
  const float* x        = (const float*)d_in[0];
  const int*   pos_ids  = (const int*)  d_in[1];
  const float* norm1_w  = (const float*)d_in[2];
  const float* norm2_w  = (const float*)d_in[3];
  const float* q_norm_w = (const float*)d_in[4];
  const float* k_norm_w = (const float*)d_in[5];
  const float* wq       = (const float*)d_in[6];
  const float* wk       = (const float*)d_in[7];
  const float* wv       = (const float*)d_in[8];
  const float* wo       = (const float*)d_in[9];
  const float* gate_w   = (const float*)d_in[10];
  const float* exp_gate = (const float*)d_in[11];
  const float* exp_up   = (const float*)d_in[12];
  const float* exp_down = (const float*)d_in[13];

  char* W = (char*)d_ws;
  size_t off = 0;
  auto alloc = [&](size_t bytes) -> void* {
    size_t cur = (off + 255) & ~(size_t)255;
    off = cur + bytes;
    return (void*)(W + cur);
  };
  u16*   h1_bf = (u16*)  alloc((size_t)TT * DD * 2);
  float* qf    = (float*)alloc((size_t)TT * HH * HDD * 4);
  float* kf    = (float*)alloc((size_t)TT * GG * HDD * 4);   // kf+vf contiguous
  float* vf    = (float*)alloc((size_t)TT * GG * HDD * 4);
  u16*   Qbf   = (u16*)  alloc((size_t)BB * HH * SS * HDD * 2);
  u16*   Kbf   = (u16*)  alloc((size_t)BB * GG * SS * HDD * 2);
  u16*   Vt    = (u16*)  alloc((size_t)BB * GG * HDD * SS * 2);
  u16*   ctx   = (u16*)  alloc((size_t)TT * HH * HDD * 2);
  int*   tok_idx     = (int*)  alloc((size_t)TT * 2 * 4);
  float* tok_w       = (float*)alloc((size_t)TT * 2 * 4);
  int*   counts      = (int*)  alloc(EE * 4);
  int*   bases       = (int*)  alloc(EE * 4);
  int*   entry_token = (int*)  alloc((size_t)TT * 2 * 4);
  float* entry_w     = (float*)alloc((size_t)TT * 2 * 4);
  int*   entry_of    = (int*)  alloc((size_t)TT * 2 * 4);
  u16*   he          = (u16*)  alloc((size_t)TT * 2 * HID * 2);
  float* dout        = (float*)alloc((size_t)TT * 2 * DD * 4);
  // reuse freed regions: x2 over qf (16MB), h2 over kf+vf (8MB)
  float* x2    = qf;
  u16*   h2_bf = (u16*)kf;

  // 1) rmsnorm1 -> bf16
  rmsnorm_kernel<<<TT, 256, 0, stream>>>(x, norm1_w, h1_bf);
  // 2) QKV projections (WMMA)
  gemm_aw<<<dim3((HH*HDD)/BN, TT/BM), 256, 0, stream>>>(h1_bf, wq, qf, TT, HH*HDD, DD, nullptr);
  gemm_aw<<<dim3((GG*HDD)/BN, TT/BM), 256, 0, stream>>>(h1_bf, wk, kf, TT, GG*HDD, DD, nullptr);
  gemm_aw<<<dim3((GG*HDD)/BN, TT/BM), 256, 0, stream>>>(h1_bf, wv, vf, TT, GG*HDD, DD, nullptr);
  // 3) q/k rmsnorm + rope, pack bf16 (V transposed)
  rope_pack_kernel<<<dim3(HH + 2*GG, TT), 32, 0, stream>>>(qf, kf, vf, q_norm_w, k_norm_w,
                                                           pos_ids, Qbf, Kbf, Vt);
  // 4) causal flash attention (WMMA)
  attn_kernel<<<dim3(SS/128, BB*HH), 256, 0, stream>>>(Qbf, Kbf, Vt, ctx);
  // 5) out projection + residual
  gemm_aw<<<dim3(DD/BN, TT/BM), 256, 0, stream>>>(ctx, wo, x2, TT, DD, HH*HDD, x);
  // 6) rmsnorm2 -> bf16
  rmsnorm_kernel<<<TT, 256, 0, stream>>>(x2, norm2_w, h2_bf);
  // 7-9) routing: logits/top2, bases, deterministic expert lists
  zero_counts_kernel<<<1, 32, 0, stream>>>(counts);
  router_kernel<<<TT/8, 256, 0, stream>>>(h2_bf, gate_w, tok_idx, tok_w, counts);
  bases_kernel<<<1, 32, 0, stream>>>(counts, bases);
  build_lists_kernel<<<EE, 256, 0, stream>>>(tok_idx, tok_w, bases, entry_token, entry_w, entry_of);
  // 10) gate/up GEMM + SiLU*up (gathered rows, WMMA)
  gemm_gateup<<<dim3(HID/BN, TT/BM, EE), 256, 0, stream>>>(h2_bf, exp_gate, exp_up, he,
                                                           entry_token, bases, counts);
  // 11) down GEMM, scaled by routing weight
  gemm_down<<<dim3(DD/BN, TT/BM, EE), 256, 0, stream>>>(he, exp_down, dout, entry_w, bases, counts);
  // 12) combine + residual
  combine_kernel<<<TT, 256, 0, stream>>>(x2, dout, entry_of, (float*)d_out);
}
